// MinimalStreamingDurationHeadV1G_979252544163
// MI455X (gfx1250) — matrix operand order repaired and checked by reading, not compiled, not run on hardware
//
#include <hip/hip_runtime.h>
#include <math.h>

// ---------------------------------------------------------------------------
// Problem constants (match reference)
// ---------------------------------------------------------------------------
#define BB 8
#define TT 4096
#define DD 512
#define BT (BB * TT)          // 32768 tokens
#define BD (BB * DD)          // 4096
#define DECAY 0.95f
#define COARSE_SCALE 0.2f
#define RES_SCALE 0.35f
#define MAX_LS 1.2f
#define MAX_SIL 0.35f
#define COLD_RUNS 2.0f

// GEMM blocking
#define MB 4                  // 16-token M-tiles per block
#define TOK (MB * 16)         // 64 tokens per block
#define ATILE (16 * 32 * 16)  // bf16 elements per packed tile (16 ktiles x 32 lanes x 16)

typedef __attribute__((ext_vector_type(16))) __bf16 v16bf;
typedef __attribute__((ext_vector_type(8)))  float  v8f;

__device__ __forceinline__ float gelu_f(float x) {
    // exact gelu (approximate=False): x * 0.5 * (1 + erf(x / sqrt(2)))
    return 0.5f * x * (1.0f + erff(x * 0.70710678118654752440f));
}
__device__ __forceinline__ float clip01i(int v) {
    float f = (float)v;
    return fminf(fmaxf(f, 0.0f), 1.0f);
}

// ---------------------------------------------------------------------------
// Kernel A1: spk_ctx[b][d] = tanh(spk_embed[b] . spk_W[:,d] + spk_b[d])
// ---------------------------------------------------------------------------
__global__ void spkctx_kernel(const float* __restrict__ spkE,
                              const float* __restrict__ spkW,
                              const float* __restrict__ spkB,
                              float* __restrict__ wsSpkCtx) {
    int idx = blockIdx.x * blockDim.x + threadIdx.x;   // B*D threads
    if (idx >= BD) return;
    int b = idx >> 9, d = idx & (DD - 1);
    float s = spkB[d];
    for (int k = 0; k < DD; ++k)
        s = fmaf(spkE[b * DD + k], spkW[k * DD + d], s);
    wsSpkCtx[idx] = tanhf(s);
}

// ---------------------------------------------------------------------------
// Kernel A2: coarse_scalar[b] = 0.2*tanh(gelu([spk_ctx,grc] @ W1 + b1) @ W2 + b2)
// one block per batch
// ---------------------------------------------------------------------------
__global__ void coarse_kernel(const float* __restrict__ wsSpkCtx,
                              const float* __restrict__ grate,
                              const float* __restrict__ W1,
                              const float* __restrict__ b1,
                              const float* __restrict__ W2,
                              const float* __restrict__ b2,
                              float* __restrict__ wsCoarse) {
    __shared__ float redc[256];
    int b = blockIdx.x, tid = threadIdx.x;
    float grc = grate[b];
    float total = 0.0f;
    for (int d = tid; d < DD; d += 256) {
        float s = b1[d];
        for (int k = 0; k < DD; ++k)
            s = fmaf(wsSpkCtx[b * DD + k], W1[k * DD + d], s);
        s = fmaf(grc, W1[DD * DD + d], s);   // last row of (D+1, D)
        total = fmaf(gelu_f(s), W2[d], total);
    }
    redc[tid] = total;
    __syncthreads();
    for (int off = 128; off > 0; off >>= 1) {
        if (tid < off) redc[tid] += redc[tid + off];
        __syncthreads();
    }
    if (tid == 0) wsCoarse[b] = COARSE_SCALE * tanhf(redc[0] + b2[0]);
}

// ---------------------------------------------------------------------------
// Kernel A3: spk_proj[b][n] = spk_ctx[b] @ Ws[:,n], Ws = res_W1 rows [D, 2D)
// ---------------------------------------------------------------------------
__global__ void spkproj_kernel(const float* __restrict__ wsSpkCtx,
                               const float* __restrict__ resW1,
                               float* __restrict__ wsSpkProj) {
    int idx = blockIdx.x * blockDim.x + threadIdx.x;
    if (idx >= BD) return;
    int b = idx >> 9, n = idx & (DD - 1);
    float s = 0.0f;
    for (int k = 0; k < DD; ++k)
        s = fmaf(wsSpkCtx[b * DD + k], resW1[(DD + k) * DD + n], s);
    wsSpkProj[idx] = s;
}

// ---------------------------------------------------------------------------
// Kernel P: pack Wq (= res_W1 rows [0, D)) f32 -> bf16 in WMMA B-fragment order.
// B (32x16, 16-bit) lane layout: lane n in [0,16): element j -> K=j;
//                                lane n+16:        element j -> K=16+j   (contiguous k order, same as A)
// Packed address: ((ntile*16 + ktile)*32 + lane)*16 + j  (bf16 elements)
// ---------------------------------------------------------------------------
__global__ void pack_kernel(const float* __restrict__ resW1,
                            __bf16* __restrict__ wsWq) {
    int blk = blockIdx.x;          // 0..511 = ntile*16 + ktile
    int ntile = blk >> 4, kt = blk & 15;
    int tid = threadIdx.x;         // 512 threads
    int lane = tid & 31, j = tid >> 5;     // j in 0..15
    int nl = lane & 15, khalf = lane >> 4;
    int k = kt * 32 + khalf * 16 + j;
    int n = ntile * 16 + nl;
    wsWq[(blk * 32 + lane) * 16 + j] = (__bf16)resW1[k * DD + n];
}

// ---------------------------------------------------------------------------
// Kernel S: two-level per-batch scan. 256 threads (8 waves) per batch.
// Each wave owns a 512-token segment. Phase 1: per-wave segment affine
// composition + speech sum. Phase 2: thread 0 combines the 8 segment
// summaries into per-segment carries. Phase 3: each wave replays its segment
// with the right carry and writes all per-token terms.
// ---------------------------------------------------------------------------
__global__ __launch_bounds__(256) void scan_kernel(
        const float* __restrict__ la_g,
        const int*   __restrict__ um,
        const int*   __restrict__ sm,
        const int*   __restrict__ silm,
        const float* __restrict__ stab,
        const float* __restrict__ grate,
        const float* __restrict__ lre,
        const float* __restrict__ wsCoarse,
        float* __restrict__ wsRate,
        float* __restrict__ wsSil,
        float* __restrict__ wsGt,
        float* __restrict__ wsRg,
        float* __restrict__ wsSp,
        float* __restrict__ wsSc) {
    __shared__ float segA[8], segB[8], segS[8];
    __shared__ float carryIn[8], sumIn[8];
    const int SEG = TT / 8;            // 512 tokens per wave
    int b = blockIdx.x;
    int tid = threadIdx.x;
    int lane = tid & 31, wave = tid >> 5;
    int base = b * TT + wave * SEG;

    // ---- Phase 1: segment summary (no outputs) ----
    float Aseg = 1.0f, Bseg = 0.0f, Sseg = 0.0f;
    for (int c = 0; c < SEG; c += 32) {
        int t = base + c + lane;
        float la = la_g[t];
        float mask = clip01i(um[t]);
        float sealed = clip01i(sm[t]);
        float sil = clip01i(silm[t]) * mask;
        float commit = mask * sealed;
        float speech = commit * (1.0f - sil);
        float a  = (speech > 0.0f) ? DECAY : 1.0f;
        float bb = (speech > 0.0f) ? (1.0f - DECAY) * la : 0.0f;
        float A = a, Bv = bb, S = speech;
        for (int d = 1; d < 32; d <<= 1) {
            float Au = __shfl_up(A, (unsigned)d, 32);
            float Bu = __shfl_up(Bv, (unsigned)d, 32);
            float Su = __shfl_up(S, (unsigned)d, 32);
            if (lane >= d) { Bv = fmaf(A, Bu, Bv); A = A * Au; S += Su; }
        }
        float Atot = __shfl(A, 31, 32);
        float Btot = __shfl(Bv, 31, 32);
        float Stot = __shfl(S, 31, 32);
        // seg := chunk ∘ seg
        Bseg = fmaf(Atot, Bseg, Btot);
        Aseg = Atot * Aseg;
        Sseg += Stot;
    }
    if (lane == 0) { segA[wave] = Aseg; segB[wave] = Bseg; segS[wave] = Sseg; }
    __syncthreads();

    // ---- Phase 2: exclusive combine of segment summaries ----
    if (tid == 0) {
        float Ap = 1.0f, Bp = 0.0f, Sp = 0.0f;
        float init = lre[b];
        for (int w = 0; w < 8; ++w) {
            carryIn[w] = fmaf(Ap, init, Bp);
            sumIn[w]   = Sp;
            // prefix := seg_w ∘ prefix
            Bp = fmaf(segA[w], Bp, segB[w]);
            Ap = segA[w] * Ap;
            Sp += segS[w];
        }
    }
    __syncthreads();

    // ---- Phase 3: replay segment with carry, write outputs ----
    float carry = carryIn[wave];
    float runSum = sumIn[wave];
    float grc = grate[b];
    float cs = wsCoarse[b];
    for (int c = 0; c < SEG; c += 32) {
        int t = base + c + lane;
        float la = la_g[t];
        float mask = clip01i(um[t]);
        float sealed = clip01i(sm[t]);
        float sil = clip01i(silm[t]) * mask;
        float stability = fminf(fmaxf(stab[t], 0.0f), 1.0f) * mask;
        float commit = mask * sealed;
        float silc = commit * sil;
        float speech = commit * (1.0f - sil);
        float a  = (speech > 0.0f) ? DECAY : 1.0f;
        float bb = (speech > 0.0f) ? (1.0f - DECAY) * la : 0.0f;
        float A = a, Bv = bb, S = speech;
        for (int d = 1; d < 32; d <<= 1) {
            float Au = __shfl_up(A, (unsigned)d, 32);
            float Bu = __shfl_up(Bv, (unsigned)d, 32);
            float Su = __shfl_up(S, (unsigned)d, 32);
            if (lane >= d) { Bv = fmaf(A, Bu, Bv); A = A * Au; S += Su; }
        }
        float Aex = __shfl_up(A, 1u, 32);
        float Bex = __shfl_up(Bv, 1u, 32);
        float Sex = __shfl_up(S, 1u, 32);
        if (lane == 0) { Aex = 1.0f; Bex = 0.0f; Sex = 0.0f; }
        float rate = fmaf(Aex, carry, Bex);     // rate BEFORE this step
        float prefix = runSum + Sex;            // exclusive speech prefix
        float Atot = __shfl(A, 31, 32);
        float Btot = __shfl(Bv, 31, 32);
        float Stot = __shfl(S, 31, 32);
        carry = fmaf(Atot, carry, Btot);
        runSum += Stot;

        float gsa = (grc - rate) * commit;
        float gt = (gsa + cs * commit) * commit;
        float cold = fminf(fmaxf(prefix / COLD_RUNS, 0.0f), 1.0f);
        float sg = fminf(fmaxf(fmaxf(expf(la), 1.0f) - 1.0f, 0.0f), 1.0f);
        float rg = cold * sg * stability * speech;

        wsRate[t] = rate;
        wsSil[t]  = sil;
        wsGt[t]   = gt;
        wsRg[t]   = rg;
        wsSp[t]   = speech;
        wsSc[t]   = silc;
    }
}

// ---------------------------------------------------------------------------
// Kernel C: fused query-build + (query @ Wq) via bf16 WMMA + epilogue.
// Block = 128 threads (4 waves), MB=4 M-tiles (64 tokens) per block so each
// B-fragment load from L2 feeds 4 WMMAs. A-fragments are deliberately
// RE-READ from LDS inside the N-tile loop (opaque offset defeats LICM):
// this keeps the wave under 256 VGPRs, removes s_set_vgpr_msb churn, lets
// the scheduler multi-buffer B loads, and allows ~4 blocks/WGP occupancy
// (66 KB LDS each) so wave switching hides B latency.
// A-fragment LDS layout (contiguous k as in packed B, 16-bit A 16x32):
//   lane m in [0,16): elements 0..15 -> K=0..15
//   lane m+16:        elements 0..15 -> K=16..31
// ---------------------------------------------------------------------------
__global__ __launch_bounds__(128) void fused_kernel(
        const int*   __restrict__ units,
        const float* __restrict__ la_g,
        const int*   __restrict__ um,
        const float* __restrict__ sep,
        const float* __restrict__ edge,
        const float* __restrict__ emb,
        const float* __restrict__ featW,
        const float* __restrict__ featB,
        const float* __restrict__ resW1,
        const float* __restrict__ resB1,
        const float* __restrict__ resW2,
        const float* __restrict__ resB2,
        const float* __restrict__ wsRate,
        const float* __restrict__ wsSil,
        const float* __restrict__ wsGt,
        const float* __restrict__ wsRg,
        const float* __restrict__ wsSp,
        const float* __restrict__ wsSc,
        const float* __restrict__ wsSpkProj,
        const __bf16* __restrict__ wsWq,
        float* __restrict__ out) {
    extern __shared__ __align__(32) __bf16 aT[];   // MB * ATILE bf16 = 64 KB
    __shared__ float s_la[TOK], s_rate[TOK], s_sil[TOK], s_sep[TOK], s_edge[TOK];
    __shared__ float s_mask[TOK], s_gt[TOK], red[TOK];
    __shared__ int   s_unit[TOK];

    int tid = threadIdx.x;
    int lane = tid & 31, wave = tid >> 5;
    int tok0 = blockIdx.x * TOK;
    int b = tok0 / TT;                 // 4096 % 64 == 0 -> uniform per block

    if (tid < TOK) {
        int t = tok0 + tid;
        s_unit[tid] = units[t];
        s_la[tid]   = la_g[t];
        s_rate[tid] = wsRate[t];
        s_sil[tid]  = wsSil[t];
        s_sep[tid]  = sep[t];
        s_edge[tid] = edge[t];
        s_mask[tid] = clip01i(um[t]);
        s_gt[tid]   = wsGt[t];
        red[tid]    = 0.0f;
    }
    __syncthreads();

    // Build query tiles (embedding gather + 5-feature matmul + gelu + mask),
    // packed bf16 directly into the WMMA A-fragment LDS layout.
    for (int idx = tid; idx < TOK * DD; idx += 128) {
        int m = idx >> 9, k = idx & (DD - 1);
        float q = emb[s_unit[m] * DD + k] + featB[k];
        q = fmaf(s_la[m],   featW[0 * DD + k], q);
        q = fmaf(s_rate[m], featW[1 * DD + k], q);
        q = fmaf(s_sil[m],  featW[2 * DD + k], q);
        q = fmaf(s_sep[m],  featW[3 * DD + k], q);
        q = fmaf(s_edge[m], featW[4 * DD + k], q);
        q = gelu_f(q) * s_mask[m];
        int mt = m >> 4, ml = m & 15;
        int kt = k >> 5, kl = k & 31;
        int khalf = kl >> 4;
        int j = kl & 15;
        aT[mt * ATILE + (kt * 32 + (ml + 16 * khalf)) * 16 + j] = (__bf16)q;
    }
    __syncthreads();

    int nIn = lane & 15, hi = lane >> 4;
    float part[MB][8];
#pragma unroll
    for (int mt = 0; mt < MB; ++mt)
#pragma unroll
        for (int r = 0; r < 8; ++r) part[mt][r] = 0.0f;

    // Each wave: 8 N-tiles of 16 columns; K = 512 => 16 k-steps, 4 WMMA each.
    for (int i = 0; i < 8; ++i) {
        int ntile = wave * 8 + i;
        const __bf16* bbase = wsWq + (size_t)ntile * ATILE;
        if (i < 7) {
            // prefetch the whole next 16 KB B block: 4 cachelines (128 B) per lane
            const __bf16* bn = bbase + ATILE;
#pragma unroll
            for (int p = 0; p < 4; ++p)
                __builtin_prefetch(bn + lane * 256 + p * 64, 0, 1);
        }
        // Opaque zero offset: prevents hoisting the 64 LDS A-fragment loads
        // out of the N-tile loop (keeps VGPR pressure low for B pipelining).
        int aOff = 0;
        asm volatile("" : "+v"(aOff));

        v8f acc[MB];
#pragma unroll
        for (int mt = 0; mt < MB; ++mt)
            acc[mt] = (v8f){0.f, 0.f, 0.f, 0.f, 0.f, 0.f, 0.f, 0.f};

        // depth-2 software pipeline on the B operand
        v16bf bP0 = *(const v16bf*)&bbase[(0 * 32 + lane) * 16];
        v16bf bP1 = *(const v16bf*)&bbase[(1 * 32 + lane) * 16];
#pragma unroll
        for (int kt = 0; kt < 16; ++kt) {
            v16bf bcur = bP0;
            bP0 = bP1;
            if (kt < 14)
                bP1 = *(const v16bf*)&bbase[((kt + 2) * 32 + lane) * 16];
#pragma unroll
            for (int mt = 0; mt < MB; ++mt) {
                v16bf af = *(const v16bf*)&aT[mt * ATILE + (kt * 32 + lane) * 16 + aOff];
                acc[mt] = __builtin_amdgcn_wmma_f32_16x16x32_bf16(
                              false, af, false, bcur, (short)0, acc[mt], false, false);
            }
        }
        // epilogue: C/D layout -> lane l, vgpr r: M = r + 8*(l>=16), N = l&15
        int n = ntile * 16 + nIn;
        float spkp = wsSpkProj[b * DD + n];
        float wgn  = resW1[2 * DD * DD + n];   // wg = res_W1[2D] row
        float rb   = resB1[n];
        float w2   = resW2[n];
#pragma unroll
        for (int mt = 0; mt < MB; ++mt) {
#pragma unroll
            for (int r = 0; r < 8; ++r) {
                int m = mt * 16 + r + 8 * hi;
                float x = acc[mt][r] + spkp + s_gt[m] * wgn + rb;
                part[mt][r] = fmaf(gelu_f(x), w2, part[mt][r]);
            }
        }
    }
    // reduce h . res_W2 across lanes/waves into per-row sums (ds_add_f32)
#pragma unroll
    for (int mt = 0; mt < MB; ++mt)
#pragma unroll
        for (int r = 0; r < 8; ++r)
            atomicAdd(&red[mt * 16 + r + 8 * hi], part[mt][r]);
    __syncthreads();

    if (tid < TOK) {
        int t = tok0 + tid;
        float resid = RES_SCALE * tanhf(red[tid] + resB2[0]);
        resid *= wsRg[t];
        float gt = s_gt[tid];
        float ps   = fminf(fmaxf(gt + resid, -MAX_LS), MAX_LS) * wsSp[t];
        float psil = fminf(fmaxf(gt, -MAX_SIL), MAX_SIL) * wsSc[t];
        out[t] = ps + psil;
    }
}

// ---------------------------------------------------------------------------
// Launcher
// ---------------------------------------------------------------------------
extern "C" void kernel_launch(void* const* d_in, const int* in_sizes, int n_in,
                              void* d_out, int out_size, void* d_ws, size_t ws_size,
                              hipStream_t stream) {
    (void)in_sizes; (void)n_in; (void)out_size; (void)ws_size;
    const int*   content_units = (const int*)d_in[0];
    const float* log_anchor    = (const float*)d_in[1];
    const int*   unit_mask     = (const int*)d_in[2];
    const int*   sealed_mask   = (const int*)d_in[3];
    const int*   silence_mask  = (const int*)d_in[4];
    const float* sep_hint      = (const float*)d_in[5];
    const float* edge_cue      = (const float*)d_in[6];
    const float* run_stability = (const float*)d_in[7];
    const float* global_rate   = (const float*)d_in[8];
    const float* local_rate_ema= (const float*)d_in[9];
    const float* spk_embed     = (const float*)d_in[10];
    const float* emb_table     = (const float*)d_in[11];
    const float* feat_W        = (const float*)d_in[12];
    const float* feat_b        = (const float*)d_in[13];
    const float* spk_W         = (const float*)d_in[14];
    const float* spk_b         = (const float*)d_in[15];
    const float* coarse_W1     = (const float*)d_in[16];
    const float* coarse_b1     = (const float*)d_in[17];
    const float* coarse_W2     = (const float*)d_in[18];
    const float* coarse_b2     = (const float*)d_in[19];
    const float* res_W1        = (const float*)d_in[20];
    const float* res_b1        = (const float*)d_in[21];
    const float* res_W2        = (const float*)d_in[22];
    const float* res_b2        = (const float*)d_in[23];
    // d_in[24] src_rate_init: unused by the reference computation
    float* out = (float*)d_out;

    // workspace layout
    float* ws        = (float*)d_ws;
    float* wsRate    = ws;                 // BT
    float* wsSil     = wsRate + BT;        // BT
    float* wsGt      = wsSil + BT;         // BT
    float* wsRg      = wsGt + BT;          // BT
    float* wsSp      = wsRg + BT;          // BT
    float* wsSc      = wsSp + BT;          // BT
    float* wsSpkCtx  = wsSc + BT;          // BD
    float* wsSpkProj = wsSpkCtx + BD;      // BD
    float* wsCoarse  = wsSpkProj + BD;     // 8 (pad 16)
    __bf16* wsWq     = (__bf16*)(wsCoarse + 16);   // 512*512 bf16

    spkctx_kernel<<<BD / 256, 256, 0, stream>>>(spk_embed, spk_W, spk_b, wsSpkCtx);
    coarse_kernel<<<BB, 256, 0, stream>>>(wsSpkCtx, global_rate, coarse_W1, coarse_b1,
                                          coarse_W2, coarse_b2, wsCoarse);
    spkproj_kernel<<<BD / 256, 256, 0, stream>>>(wsSpkCtx, res_W1, wsSpkProj);
    pack_kernel<<<512, 512, 0, stream>>>(res_W1, wsWq);
    scan_kernel<<<BB, 256, 0, stream>>>(log_anchor, unit_mask, sealed_mask, silence_mask,
                                        run_stability, global_rate, local_rate_ema,
                                        wsCoarse, wsRate, wsSil, wsGt, wsRg, wsSp, wsSc);
    size_t dynLds = (size_t)MB * ATILE * sizeof(__bf16);   // 64 KB
    fused_kernel<<<BT / TOK, 128, dynLds, stream>>>(content_units, log_anchor, unit_mask,
                                                    sep_hint, edge_cue, emb_table, feat_W, feat_b,
                                                    res_W1, res_b1, res_W2, res_b2,
                                                    wsRate, wsSil, wsGt, wsRg, wsSp, wsSc,
                                                    wsSpkProj, wsWq, out);
}
